// IMPNLayer_19026705122046
// MI455X (gfx1250) — compile-verified, matrix operand order checked by reading, not run on hardware
//
#include <hip/hip_runtime.h>
#include <math.h>

// ---------------- problem constants (from reference) ----------------
#define BATCH 2
#define NA 384
#define NB 384
#define D_A 128
#define D_B 64
#define DMSG 256
#define HEADS 8
#define DH 32
#define DFF 1024
#define MROWS (BATCH * NA)   // 768 rows when batch is folded into M

typedef __attribute__((ext_vector_type(16))) __bf16 v16bf;
typedef __attribute__((ext_vector_type(8)))  float  v8f;

__device__ __forceinline__ float gelu_exact(float x) {
  return 0.5f * x * (1.0f + erff(x * 0.7071067811865475f));
}
__device__ __forceinline__ float warp_sum(float v) {
#pragma unroll
  for (int o = 16; o > 0; o >>= 1) v += __shfl_xor(v, o, 32);
  return v;
}
__device__ __forceinline__ float warp_max(float v) {
#pragma unroll
  for (int o = 16; o > 0; o >>= 1) v = fmaxf(v, __shfl_xor(v, o, 32));
  return v;
}

// ---------------------------------------------------------------------------
// bf16 WMMA 16x16x32 tiled GEMM:  C = epilogue(alpha * A(MxK) @ B(KxN) + bias)
//  - BT:  B stored as [N,K] row-major (used for q@k^T and a@b^T affinity)
//  - ACT: 1 -> exact GELU epilogue
//  - ADD: C += result (residual accumulate)
//  - two-level batch: z -> (bb = z/batch2, hh = z%batch2) with per-level strides
// Fragment layouts per CDNA5 ISA 7.12.2 (16-bit A 16x32):
//   lane L: row = L&15, half = L>>4 ; VGPR v holds K pair
//   k = (v>>2)*16 + half*8 + (v&3)*2 ; B symmetric with lane = output column.
// K-pairs (k, k+1) are contiguous and 8B-aligned for A (always) and for B when
// BT=1, so those feed from float2 (global_load_b64) into v_cvt_pk_bf16_f32.
// One wave computes one 16x16 C tile; 4 waves / 128-thread block.
// ---------------------------------------------------------------------------
template <int ACT, bool ADD, bool BT>
__global__ __launch_bounds__(128) void gemm_wmma(
    const float* __restrict__ A, int lda,
    const float* __restrict__ Bm, int ldb,
    const float* __restrict__ bias,
    float* __restrict__ C, int ldc,
    int M, int N, int K, float alpha, int batch2,
    long long sAb, long long sAh, long long sBb, long long sBh,
    long long sCb, long long sCh) {
  const int wave = threadIdx.x >> 5;
  const int lane = threadIdx.x & 31;
  const int ntiles_n = N >> 4;
  const int ntiles = (M >> 4) * ntiles_n;
  const int tile = blockIdx.x * 4 + wave;
  if (tile >= ntiles) return;               // wave-uniform: EXEC stays all-ones
  const int mt = tile / ntiles_n;
  const int nt = tile % ntiles_n;
  const int z = blockIdx.z;
  const int bb = z / batch2, hh = z % batch2;
  A  += bb * sAb + hh * sAh;
  Bm += bb * sBb + hh * sBh;
  C  += bb * sCb + hh * sCh;

  const int fr = lane & 15;   // A row / B col within the tile
  const int hf = lane >> 4;

  // hoisted per-lane base pointers
  const float* arow = A + (long long)(mt * 16 + fr) * lda;
  const float* brow = BT ? (Bm + (long long)(nt * 16 + fr) * ldb)
                         : (Bm + nt * 16 + fr);

  v8f acc = {};
  for (int kb = 0; kb < K; kb += 32) {
    v16bf afrag, bfrag;
#pragma unroll
    for (int v = 0; v < 8; ++v) {
      const int k0 = ((v >> 2) << 4) + (hf << 3) + ((v & 3) << 1);
      const float2 a2 = *(const float2*)(arow + kb + k0);
      afrag[2 * v]     = (__bf16)a2.x;
      afrag[2 * v + 1] = (__bf16)a2.y;
      if (BT) {
        const float2 b2 = *(const float2*)(brow + kb + k0);
        bfrag[2 * v]     = (__bf16)b2.x;
        bfrag[2 * v + 1] = (__bf16)b2.y;
      } else {
        bfrag[2 * v]     = (__bf16)brow[(long long)(kb + k0) * ldb];
        bfrag[2 * v + 1] = (__bf16)brow[(long long)(kb + k0 + 1) * ldb];
      }
    }
    acc = __builtin_amdgcn_wmma_f32_16x16x32_bf16(false, afrag, false, bfrag,
                                                  (short)0, acc, false, false);
  }

  // ---- epilogue: single bias load, running C pointer (no per-r branches) ----
  const int n = nt * 16 + fr;
  float bvn = 0.0f;
  if (bias) bvn = bias[n];
  float* crow = C + (long long)(mt * 16 + hf * 8) * ldc + n;
#pragma unroll
  for (int r = 0; r < 8; ++r) {
    float val = acc[r] * alpha + bvn;
    if (ACT == 1) val = gelu_exact(val);
    if (ADD) val += *crow;
    *crow = val;
    crow += ldc;
  }
}

// ------------------- row-wise VALU kernels (wave32 per row) ----------------

// softmax over 384 columns, in place. rows = B*H*Nq.
__global__ __launch_bounds__(256) void softmax_inplace(float* S, int rows) {
  const int wave = threadIdx.x >> 5, lane = threadIdx.x & 31;
  const int row = blockIdx.x * 8 + wave;
  if (row >= rows) return;
  float* p = S + (long long)row * 384;
  float v[12];
  float mx = -3.4e38f;
#pragma unroll
  for (int t = 0; t < 12; ++t) { v[t] = p[lane + t * 32]; mx = fmaxf(mx, v[t]); }
  mx = warp_max(mx);
  float s = 0.f;
#pragma unroll
  for (int t = 0; t < 12; ++t) { v[t] = expf(v[t] - mx); s += v[t]; }
  s = warp_sum(s);
  const float inv = 1.0f / s;
#pragma unroll
  for (int t = 0; t < 12; ++t) p[lane + t * 32] = v[t] * inv;
}

// per (b,i): mean over heads of softmax(scores[b,h,i,:]) -> out[b,i,:]
__global__ __launch_bounds__(256) void softmax_mean_heads(
    const float* __restrict__ S, float* __restrict__ out, int rows /*B*Nq*/) {
  const int wave = threadIdx.x >> 5, lane = threadIdx.x & 31;
  const int row = blockIdx.x * 8 + wave;
  if (row >= rows) return;
  const int bb = row / NA, i = row % NA;
  float acc[12];
#pragma unroll
  for (int t = 0; t < 12; ++t) acc[t] = 0.f;
  for (int h = 0; h < HEADS; ++h) {
    const float* p = S + ((long long)(bb * HEADS + h) * NA + i) * (long long)NB;
    float v[12];
    float mx = -3.4e38f;
#pragma unroll
    for (int t = 0; t < 12; ++t) { v[t] = p[lane + t * 32]; mx = fmaxf(mx, v[t]); }
    mx = warp_max(mx);
    float s = 0.f;
#pragma unroll
    for (int t = 0; t < 12; ++t) { v[t] = expf(v[t] - mx); s += v[t]; }
    s = warp_sum(s);
    const float inv = (1.0f / s) * (1.0f / HEADS);
#pragma unroll
    for (int t = 0; t < 12; ++t) acc[t] += v[t] * inv;
  }
  float* o = out + ((long long)bb * NA + i) * (long long)NB;
#pragma unroll
  for (int t = 0; t < 12; ++t) o[lane + t * 32] = acc[t];
}

// LayerNorm over 256 columns.
__global__ __launch_bounds__(256) void layernorm_rows(
    const float* __restrict__ x, const float* __restrict__ g,
    const float* __restrict__ b, float* __restrict__ y, int rows) {
  const int wave = threadIdx.x >> 5, lane = threadIdx.x & 31;
  const int row = blockIdx.x * 8 + wave;
  if (row >= rows) return;
  const float* xr = x + (long long)row * DMSG;
  float v[8];
  float s = 0.f;
#pragma unroll
  for (int t = 0; t < 8; ++t) { v[t] = xr[lane + t * 32]; s += v[t]; }
  const float mean = warp_sum(s) * (1.0f / DMSG);
  float var = 0.f;
#pragma unroll
  for (int t = 0; t < 8; ++t) { const float d = v[t] - mean; var += d * d; }
  var = warp_sum(var) * (1.0f / DMSG);
  const float inv = rsqrtf(var + 1e-5f);
  float* yr = y + (long long)row * DMSG;
#pragma unroll
  for (int t = 0; t < 8; ++t) {
    const int c = lane + t * 32;
    yr[c] = (v[t] - mean) * inv * g[c] + b[c];
  }
}

// y = x / (||x||_2 + 1e-8) over 256 columns.
__global__ __launch_bounds__(256) void l2norm_rows(
    const float* __restrict__ x, float* __restrict__ y, int rows) {
  const int wave = threadIdx.x >> 5, lane = threadIdx.x & 31;
  const int row = blockIdx.x * 8 + wave;
  if (row >= rows) return;
  const float* xr = x + (long long)row * DMSG;
  float v[8];
  float s = 0.f;
#pragma unroll
  for (int t = 0; t < 8; ++t) { v[t] = xr[lane + t * 32]; s += v[t] * v[t]; }
  s = warp_sum(s);
  const float inv = 1.0f / (sqrtf(s) + 1e-8f);
  float* yr = y + (long long)row * DMSG;
#pragma unroll
  for (int t = 0; t < 8; ++t) yr[lane + t * 32] = v[t] * inv;
}

// gate[b,i,j] = sigmoid( sum_d gelu(ta[b,i,d]+tb[b,j,d]+aff*w1r[d]+b1[d]) * w2[d] + b2 )
__global__ __launch_bounds__(256) void edge_gate_kernel(
    const float* __restrict__ ta, const float* __restrict__ tb,
    const float* __restrict__ aff,
    const float* __restrict__ w1r, const float* __restrict__ b1,
    const float* __restrict__ w2, const float* __restrict__ b2,
    float* __restrict__ gate) {
  __shared__ float sw1[DMSG], sb1[DMSG], sw2[DMSG];
  sw1[threadIdx.x] = w1r[threadIdx.x];
  sb1[threadIdx.x] = b1[threadIdx.x];
  sw2[threadIdx.x] = w2[threadIdx.x];
  __syncthreads();
  const int wave = threadIdx.x >> 5, lane = threadIdx.x & 31;
  const long long pair = (long long)blockIdx.x * 8 + wave;
  if (pair >= (long long)BATCH * NA * NB) return;
  const int b = (int)(pair / (NA * NB));
  const int rem = (int)(pair % (NA * NB));
  const int i = rem / NB, j = rem % NB;
  const float av = aff[pair];
  const float* tai = ta + ((long long)b * NA + i) * DMSG;
  const float* tbj = tb + ((long long)b * NB + j) * DMSG;
  float dot = 0.f;
#pragma unroll
  for (int t = 0; t < 8; ++t) {
    const int d = lane + t * 32;
    const float h = tai[d] + tbj[d] + av * sw1[d] + sb1[d];
    dot += gelu_exact(h) * sw2[d];
  }
  dot = warp_sum(dot);
  if (lane == 0) gate[pair] = 1.0f / (1.0f + expf(-(dot + b2[0])));
}

// wab_o[b,i,j] = wab_mean[b,i,j]*gate[b,i,j];  wba_o[b,j,i] = wba_mean[b,j,i]*gate[b,i,j]
__global__ __launch_bounds__(256) void apply_gate_kernel(
    const float* __restrict__ wabm, const float* __restrict__ wbam,
    const float* __restrict__ gate, float* __restrict__ wab_o,
    float* __restrict__ wba_o) {
  const long long idx = (long long)blockIdx.x * 256 + threadIdx.x;
  if (idx >= (long long)BATCH * NA * NB) return;
  const int b = (int)(idx / (NA * NB));
  const int rem = (int)(idx % (NA * NB));
  const int i = rem / NB, j = rem % NB;
  const float g = gate[idx];
  wab_o[idx] = wabm[idx] * g;
  const long long wi = ((long long)b * NB + j) * NA + i;
  wba_o[wi] = wbam[wi] * g;
}

// keep top-32 per 384-column row, zero the rest (ties -> lowest index, like lax.top_k)
__global__ __launch_bounds__(256) void topk_mask_k32(float* __restrict__ W, int rows) {
  const int wave = threadIdx.x >> 5, lane = threadIdx.x & 31;
  const int row = blockIdx.x * 8 + wave;
  if (row >= rows) return;
  float* p = W + (long long)row * 384;
  float vals[12];
#pragma unroll
  for (int t = 0; t < 12; ++t) vals[t] = p[lane + t * 32];
  unsigned keep = 0;
  for (int it = 0; it < 32; ++it) {
    float bv = -3.4e38f;
    int bc = 1 << 30;
#pragma unroll
    for (int t = 0; t < 12; ++t) {
      const int c = lane + t * 32;
      if (vals[t] > bv) { bv = vals[t]; bc = c; }
    }
#pragma unroll
    for (int o = 16; o > 0; o >>= 1) {
      const float ov = __shfl_xor(bv, o, 32);
      const int oc = __shfl_xor(bc, o, 32);
      if (ov > bv || (ov == bv && oc < bc)) { bv = ov; bc = oc; }
    }
    if ((bc & 31) == lane) {
      const int t = bc >> 5;
      vals[t] = -3.4e38f;
      keep |= 1u << t;
    }
  }
#pragma unroll
  for (int t = 0; t < 12; ++t)
    if (!((keep >> t) & 1u)) p[lane + t * 32] = 0.0f;
}

// ---------------------------------------------------------------------------

#define GEMM_LAUNCH(ACT, ADD, BT, Aptr, lda, Bptr, ldb, biasptr, Cptr, ldc, M, N, \
                    K, alpha, Z, B2, sAb, sAh, sBb, sBh, sCb, sCh)                \
  gemm_wmma<ACT, ADD, BT>                                                         \
      <<<dim3((unsigned)((((M) / 16) * ((N) / 16) + 3) / 4), 1, (unsigned)(Z)),   \
         128, 0, stream>>>((Aptr), (lda), (Bptr), (ldb), (biasptr), (Cptr),       \
                           (ldc), (M), (N), (K), (alpha), (B2),                   \
                           (long long)(sAb), (long long)(sAh), (long long)(sBb),  \
                           (long long)(sBh), (long long)(sCb), (long long)(sCh))

extern "C" void kernel_launch(void* const* d_in, const int* in_sizes, int n_in,
                              void* d_out, int out_size, void* d_ws,
                              size_t ws_size, hipStream_t stream) {
  (void)in_sizes; (void)n_in; (void)out_size; (void)ws_size;
  auto P = [&](int i) { return (const float*)d_in[i]; };
  // d_in layout (setup_inputs dict order, params flattened in insertion order):
  // 0:A_nodes 1:B_nodes 2:pa_w 3:pa_b 4:pb_w 5:pb_b
  // 6..13: sa q/k/v/o (w,b)  14..21: sb  22..29: ca  30..37: cb
  // 38..43: ffa lng,lnb,w1,b1,w2,b2   44..49: ffb
  // 50: edge_w1 [513,256] 51: edge_b1 52: edge_w2 [256,1] 53: edge_b2 54: topk(=32)

  // -------- workspace carve-out (~24.4 MB of f32) --------
  float* cur = (float*)d_ws;
  float* a_buf   = cur; cur += (long long)MROWS * DMSG;
  float* b_buf   = cur; cur += (long long)MROWS * DMSG;
  float* q_buf   = cur; cur += (long long)MROWS * DMSG;
  float* k_buf   = cur; cur += (long long)MROWS * DMSG;
  float* v_buf   = cur; cur += (long long)MROWS * DMSG;
  float* ao_buf  = cur; cur += (long long)MROWS * DMSG;
  float* ln_buf  = cur; cur += (long long)MROWS * DMSG;
  float* hid_buf = cur; cur += (long long)MROWS * DFF;
  float* scores  = cur; cur += (long long)BATCH * HEADS * NA * NB;
  float* wab_mean = cur; cur += (long long)BATCH * NA * NB;
  float* wba_mean = cur; cur += (long long)BATCH * NA * NB;
  float* gate_buf = cur; cur += (long long)BATCH * NA * NB;
  float* aff_buf  = cur; cur += (long long)BATCH * NA * NB;
  float* ta_buf   = cur; cur += (long long)MROWS * DMSG;
  float* tb_buf   = cur; cur += (long long)MROWS * DMSG;

  float* out_a = (float*)d_out;
  float* out_b = out_a + (long long)MROWS * DMSG;
  float* wab_o = out_b + (long long)MROWS * DMSG;
  float* wba_o = wab_o + (long long)BATCH * NA * NB;

  const float alpha_self = 1.0f / (sqrtf((float)DH) * 0.5f);
  const float alpha_cross = 1.0f / (sqrtf((float)DH) * 0.2f);

  // -------- input projections --------
  GEMM_LAUNCH(0, false, false, P(0), D_A, P(2), DMSG, P(3), a_buf, DMSG,
              MROWS, DMSG, D_A, 1.0f, 1, 1, 0, 0, 0, 0, 0, 0);
  GEMM_LAUNCH(0, false, false, P(1), D_B, P(4), DMSG, P(5), b_buf, DMSG,
              MROWS, DMSG, D_B, 1.0f, 1, 1, 0, 0, 0, 0, 0, 0);

  // -------- self attention (x += O(softmax(qk^T/s) v)) --------
  auto self_attn = [&](float* x, int base) {
    GEMM_LAUNCH(0, false, false, x, DMSG, P(base + 0), DMSG, P(base + 1), q_buf,
                DMSG, MROWS, DMSG, DMSG, 1.0f, 1, 1, 0, 0, 0, 0, 0, 0);
    GEMM_LAUNCH(0, false, false, x, DMSG, P(base + 2), DMSG, P(base + 3), k_buf,
                DMSG, MROWS, DMSG, DMSG, 1.0f, 1, 1, 0, 0, 0, 0, 0, 0);
    GEMM_LAUNCH(0, false, false, x, DMSG, P(base + 4), DMSG, P(base + 5), v_buf,
                DMSG, MROWS, DMSG, DMSG, 1.0f, 1, 1, 0, 0, 0, 0, 0, 0);
    // scores[b,h] = alpha * qh @ kh^T   (K = DH = 32 -> single WMMA step)
    GEMM_LAUNCH(0, false, true, q_buf, DMSG, k_buf, DMSG, (const float*)nullptr,
                scores, NB, NA, NB, DH, alpha_self, BATCH * HEADS, HEADS,
                (long long)NA * DMSG, DH, (long long)NB * DMSG, DH,
                (long long)HEADS * NA * NB, (long long)NA * NB);
    softmax_inplace<<<BATCH * HEADS * NA / 8, 256, 0, stream>>>(
        scores, BATCH * HEADS * NA);
    // attn_out[b,h] = P @ vh
    GEMM_LAUNCH(0, false, false, scores, NB, v_buf, DMSG, (const float*)nullptr,
                ao_buf, DMSG, NA, DH, NB, 1.0f, BATCH * HEADS, HEADS,
                (long long)HEADS * NA * NB, (long long)NA * NB,
                (long long)NB * DMSG, DH, (long long)NA * DMSG, DH);
    // x += attn_out @ Wo + bo
    GEMM_LAUNCH(0, true, false, ao_buf, DMSG, P(base + 6), DMSG, P(base + 7), x,
                DMSG, MROWS, DMSG, DMSG, 1.0f, 1, 1, 0, 0, 0, 0, 0, 0);
  };
  self_attn(a_buf, 6);   // sa
  self_attn(b_buf, 14);  // sb

  // -------- FF blocks: x += W2(gelu(W1(LN(x)))) --------
  auto ff = [&](float* x, int base) {
    layernorm_rows<<<MROWS / 8, 256, 0, stream>>>(x, P(base), P(base + 1),
                                                  ln_buf, MROWS);
    GEMM_LAUNCH(1, false, false, ln_buf, DMSG, P(base + 2), DFF, P(base + 3),
                hid_buf, DFF, MROWS, DFF, DMSG, 1.0f, 1, 1, 0, 0, 0, 0, 0, 0);
    GEMM_LAUNCH(0, true, false, hid_buf, DFF, P(base + 4), DMSG, P(base + 5), x,
                DMSG, MROWS, DMSG, DFF, 1.0f, 1, 1, 0, 0, 0, 0, 0, 0);
  };
  ff(a_buf, 38);  // ffa
  ff(b_buf, 44);  // ffb

  // -------- affinity = l2norm(a) @ l2norm(b)^T --------
  l2norm_rows<<<MROWS / 8, 256, 0, stream>>>(a_buf, q_buf, MROWS);
  l2norm_rows<<<MROWS / 8, 256, 0, stream>>>(b_buf, k_buf, MROWS);
  GEMM_LAUNCH(0, false, true, q_buf, DMSG, k_buf, DMSG, (const float*)nullptr,
              aff_buf, NB, NA, NB, DMSG, 1.0f, BATCH, 1, (long long)NA * DMSG,
              0, (long long)NB * DMSG, 0, (long long)NA * NB, 0);

  // -------- edge MLP gate --------
  GEMM_LAUNCH(0, false, false, a_buf, DMSG, P(50), DMSG, (const float*)nullptr,
              ta_buf, DMSG, MROWS, DMSG, DMSG, 1.0f, 1, 1, 0, 0, 0, 0, 0, 0);
  GEMM_LAUNCH(0, false, false, b_buf, DMSG, P(50) + DMSG * DMSG, DMSG,
              (const float*)nullptr, tb_buf, DMSG, MROWS, DMSG, DMSG, 1.0f, 1,
              1, 0, 0, 0, 0, 0, 0);
  edge_gate_kernel<<<(BATCH * NA * NB) / 8, 256, 0, stream>>>(
      ta_buf, tb_buf, aff_buf, P(50) + 2 * DMSG * DMSG, P(51), P(52), P(53),
      gate_buf);

  // -------- cross attention weights (head-mean of softmax) --------
  auto cross_w = [&](const float* qsrc, const float* ksrc, int base,
                     float* mean_out) {
    GEMM_LAUNCH(0, false, false, qsrc, DMSG, P(base + 0), DMSG, P(base + 1),
                q_buf, DMSG, MROWS, DMSG, DMSG, 1.0f, 1, 1, 0, 0, 0, 0, 0, 0);
    GEMM_LAUNCH(0, false, false, ksrc, DMSG, P(base + 2), DMSG, P(base + 3),
                k_buf, DMSG, MROWS, DMSG, DMSG, 1.0f, 1, 1, 0, 0, 0, 0, 0, 0);
    GEMM_LAUNCH(0, false, true, q_buf, DMSG, k_buf, DMSG, (const float*)nullptr,
                scores, NB, NA, NB, DH, alpha_cross, BATCH * HEADS, HEADS,
                (long long)NA * DMSG, DH, (long long)NB * DMSG, DH,
                (long long)HEADS * NA * NB, (long long)NA * NB);
    softmax_mean_heads<<<BATCH * NA / 8, 256, 0, stream>>>(scores, mean_out,
                                                           BATCH * NA);
  };
  cross_w(a_buf, b_buf, 22, wab_mean);  // ca: q from a, k from b
  cross_w(b_buf, a_buf, 30, wba_mean);  // cb: q from b, k from a

  // -------- gate multiply + top-32 mask (written straight into d_out) --------
  apply_gate_kernel<<<(BATCH * NA * NB + 255) / 256, 256, 0, stream>>>(
      wab_mean, wba_mean, gate_buf, wab_o, wba_o);
  topk_mask_k32<<<MROWS / 8, 256, 0, stream>>>(wab_o, BATCH * NA);
  topk_mask_k32<<<MROWS / 8, 256, 0, stream>>>(wba_o, BATCH * NB);

  // -------- message-passing updates --------
  // a += wab_m @ b
  GEMM_LAUNCH(0, true, false, wab_o, NB, b_buf, DMSG, (const float*)nullptr,
              a_buf, DMSG, NA, DMSG, NB, 1.0f, BATCH, 1, (long long)NA * NB, 0,
              (long long)NB * DMSG, 0, (long long)NA * DMSG, 0);
  // b += wba_m @ a   (uses updated a, as in the reference)
  GEMM_LAUNCH(0, true, false, wba_o, NA, a_buf, DMSG, (const float*)nullptr,
              b_buf, DMSG, NB, DMSG, NA, 1.0f, BATCH, 1, (long long)NA * NB, 0,
              (long long)NA * DMSG, 0, (long long)NB * DMSG, 0);

  // -------- final FF blocks --------
  ff(a_buf, 38);
  ff(b_buf, 44);

  // -------- emit a, b --------
  hipMemcpyAsync(out_a, a_buf, (size_t)MROWS * DMSG * sizeof(float),
                 hipMemcpyDeviceToDevice, stream);
  hipMemcpyAsync(out_b, b_buf, (size_t)MROWS * DMSG * sizeof(float),
                 hipMemcpyDeviceToDevice, stream);
}
